// GSC_kan_56152402428565
// MI455X (gfx1250) — matrix operand order repaired and checked by reading, not compiled, not run on hardware
//
#include <hip/hip_runtime.h>
#include <hip/hip_bf16.h>

// ---------------- CDNA5 WMMA types ----------------
typedef __attribute__((ext_vector_type(16))) __bf16 v16bf;
typedef __attribute__((ext_vector_type(8)))  __bf16 v8bf;
typedef __attribute__((ext_vector_type(8)))  float  v8f;

#define B_N   16
#define CIN   128
#define COUT  256
#define GRP   8
#define HW    4096      // 64*64
#define NPIX  65536     // B*H*W
#define K1    144
#define KP1   168       // padded K stride (bank spread, 16B aligned rows)
#define K2    288
#define KP2   296
#define HALO_C 18
#define HALO_R 4
#define NHALO  72       // 4*18
#define NWF1  (GRP * 32 * KP1)          // 43008 fused layer-1 weights (bf16)
#define NWF2  (GRP * 9 * 32 * KP2)      // 681984 fused layer-2 tap weights (bf16)

// ---------------- helpers ----------------
__device__ __forceinline__ float silu_f(float v) {
    return v / (1.0f + __expf(-v));
}

// Cox-de Boor cubic B-spline bases on uniform extended grid (h=0.4, knots -2.2..2.2)
__device__ __forceinline__ void bsplines8(float x, float bs[8]) {
    const float h = 0.4f;
    float t[12];
#pragma unroll
    for (int i = 0; i < 12; ++i) t[i] = -1.0f + (float)(i - 3) * h;
    float b0[11], b1[10], b2[9];
#pragma unroll
    for (int i = 0; i < 11; ++i) b0[i] = (x >= t[i] && x < t[i + 1]) ? 1.0f : 0.0f;
#pragma unroll
    for (int i = 0; i < 10; ++i)
        b1[i] = ((x - t[i]) * b0[i] + (t[i + 2] - x) * b0[i + 1]) * 2.5f;       // 1/h
#pragma unroll
    for (int i = 0; i < 9; ++i)
        b2[i] = ((x - t[i]) * b1[i] + (t[i + 3] - x) * b1[i + 1]) * 1.25f;      // 1/(2h)
#pragma unroll
    for (int i = 0; i < 8; ++i)
        bs[i] = ((x - t[i]) * b2[i] + (t[i + 4] - x) * b2[i + 1]) * 0.833333333f; // 1/(3h)
}

// A-matrix (16x32 bf16) fragment from row-major LDS [row][k], stride kp.
// Lane l<16: row=rb+l, K = kb..kb+15; lane>=16: K = kb+16..kb+31.
__device__ __forceinline__ v16bf load_a_frag(const __bf16* lds, int rb, int kb, int kp) {
    const int l    = threadIdx.x & 31;
    const int half = l >> 4;
    const __bf16* p = lds + (rb + (l & 15)) * kp + kb + half * 16;
    v8bf lo = *(const v8bf*)(p);
    v8bf hi = *(const v8bf*)(p + 8);
    v16bf f;
#pragma unroll
    for (int i = 0; i < 8; ++i) { f[i] = lo[i]; f[i + 8] = hi[i]; }
    return f;
}

// B-matrix (32x16 bf16) fragment: lane l<16: col=rb+l holds K=kb..kb+15;
// lane>=16: same col, K=kb+16..kb+31 (same k order as the A fragment).
__device__ __forceinline__ v16bf load_b_frag(const __bf16* lds, int rb, int kb, int kp) {
    const int l    = threadIdx.x & 31;
    const int half = l >> 4;
    const __bf16* p = lds + (rb + (l & 15)) * kp + kb + half * 16;
    v8bf lo = *(const v8bf*)(p);
    v8bf hi = *(const v8bf*)(p + 8);
    v16bf f;
#pragma unroll
    for (int i = 0; i < 8; ++i) { f[i] = lo[i]; f[i + 8] = hi[i]; }
    return f;
}

__device__ __forceinline__ v8f wmma_bf16(v16bf a, v16bf b, v8f c) {
    return __builtin_amdgcn_wmma_f32_16x16x32_bf16(false, a, false, b, (short)0, c, false, false);
}

// Async DMA: 16B global -> LDS, tracked on ASYNCcnt (gfx1250 path).
__device__ __forceinline__ void async_copy_b128(unsigned lds_off, const void* gptr) {
    asm volatile("global_load_async_to_lds_b128 %0, %1, off"
                 :: "v"(lds_off), "v"(gptr) : "memory");
}
__device__ __forceinline__ void wait_async0() {
    asm volatile("s_wait_asynccnt 0" ::: "memory");
}

// ---------------- kernel 0: zero BN accumulators ----------------
__global__ void GSC_kan_zero(float* p, int n) {
    int i = blockIdx.x * blockDim.x + threadIdx.x;
    if (i < n) p[i] = 0.0f;
}

// ---------------- kernel P: pre-fuse [base|spline] weights to bf16 ----------
// Wf1: [g][oc][KP1]; Wf2: [g][tap][oc][KP2]; zero-padded to the LDS strides.
__global__ __launch_bounds__(256) void GSC_kan_prep(const float* __restrict__ wb1,
                                                    const float* __restrict__ ws1,
                                                    const float* __restrict__ wb2,
                                                    const float* __restrict__ ws2,
                                                    __bf16* __restrict__ Wf1,
                                                    __bf16* __restrict__ Wf2) {
    int i = blockIdx.x * 256 + threadIdx.x;
    if (i < NWF1) {
        int k = i % KP1, oc_g = i / KP1;          // oc_g = g*32+oc in [0,256)
        float wv = 0.0f;
        if (k < 16)      wv = wb1[oc_g * 16 + k];
        else if (k < K1) wv = ws1[oc_g * 128 + (k - 16)];
        Wf1[i] = (__bf16)wv;
    }
    if (i < NWF2) {
        int k   = i % KP2;
        int r   = i / KP2;
        int oc  = r % 32;
        int r2  = r / 32;
        int tap = r2 % 9;
        int g   = r2 / 9;
        float wv = 0.0f;
        if (k < 32)      wv = wb2[((g * 32 + oc) * 32 + k) * 9 + tap];
        else if (k < K2) wv = ws2[((g * 32 + oc) * 256 + (k - 32)) * 9 + tap];
        Wf2[i] = (__bf16)wv;
    }
}

// ---------------- kernel 1: 1x1 grouped KAN conv (WMMA GEMM) ----------------
// grid: (NPIX/64, 8); block: 128 (4 waves, one 16-pixel N-tile each)
__global__ __launch_bounds__(128) void GSC_kan1(const float* __restrict__ x,
                                                const __bf16* __restrict__ Wf1,
                                                const float* __restrict__ bb1,
                                                __bf16* __restrict__ h1) {
    __shared__ __attribute__((aligned(16))) __bf16 Wlds[32 * KP1];
    __shared__ __attribute__((aligned(16))) __bf16 Flds[64 * KP1];
    const int g  = blockIdx.y;
    const int p0 = blockIdx.x * 64;
    const int t  = threadIdx.x;

    // async DMA staged fused weights (pure contiguous copy of 32*KP1 bf16)
    {
        const __bf16* src = Wf1 + (size_t)g * 32 * KP1;
        const unsigned ldsb = (unsigned)(size_t)(&Wlds[0]);
        for (int c = t; c < (32 * KP1 * 2) / 16; c += 128)     // 672 chunks
            async_copy_b128(ldsb + c * 16, src + c * 8);
    }
    // stage features: 64 pixels x (16 silu + 128 spline)
    {
        const int c  = t & 15;
        const int ps = t >> 4;
#pragma unroll
        for (int pj = 0; pj < 8; ++pj) {
            int n  = ps + pj * 8;
            int p  = p0 + n;
            int b  = p >> 12, hw = p & 4095;
            float v = x[(b * CIN + g * 16 + c) * HW + hw];
            Flds[n * KP1 + c] = (__bf16)silu_f(v);
            float bs[8]; bsplines8(v, bs);
#pragma unroll
            for (int q = 0; q < 8; ++q)
                Flds[n * KP1 + 16 + c * 8 + q] = (__bf16)bs[q];
        }
    }
    for (int idx = t; idx < 64 * (KP1 - K1); idx += 128) {
        int n = idx / (KP1 - K1), o = idx % (KP1 - K1);
        Flds[n * KP1 + K1 + o] = (__bf16)0.0f;
    }
    wait_async0();
    __syncthreads();

    const int wv_ = t >> 5;          // wave id -> N-tile
    const int nb_ = wv_ * 16;
    v8f acc0 = {}; v8f acc1 = {};
#pragma unroll
    for (int kt = 0; kt < 5; ++kt) {
        int kb = kt * 32;
        v16bf bf = load_b_frag(Flds, nb_, kb, KP1);
        v16bf a0 = load_a_frag(Wlds, 0,  kb, KP1);
        v16bf a1 = load_a_frag(Wlds, 16, kb, KP1);
        acc0 = wmma_bf16(a0, bf, acc0);
        acc1 = wmma_bf16(a1, bf, acc1);
    }
    // write h1 (bf16) with bias; C/D layout: lane = pixel, vgpr r = row (r / r+8 per half)
    const int l    = t & 31;
    const int half = l >> 4;
    const int p    = p0 + nb_ + (l & 15);
    const int b    = p >> 12, hw = p & 4095;
#pragma unroll
    for (int r = 0; r < 8; ++r) {
        int m0  = r + half * 8;
        int ch0 = g * 32 + m0;
        int ch1 = ch0 + 16;
        h1[(b * COUT + ch0) * HW + hw] = (__bf16)(acc0[r] + bb1[ch0]);
        h1[(b * COUT + ch1) * HW + hw] = (__bf16)(acc1[r] + bb1[ch1]);
    }
}

// ---------------- kernel 2: 3x3 grouped KAN conv (WMMA) + BN partial sums ----
// grid: (128 tiles, 8 groups, 16 batch); block: 64 (2 waves, one row each)
__global__ __launch_bounds__(64) void GSC_kan2(const __bf16* __restrict__ h1,
                                               const __bf16* __restrict__ Wf2,
                                               const float* __restrict__ bb2,
                                               float* __restrict__ h2,
                                               float* __restrict__ sums,
                                               float* __restrict__ sumsq) {
    __shared__ __attribute__((aligned(16))) __bf16 Flds[NHALO * KP2]; // 4x18 halo, 296 feats
    __shared__ __attribute__((aligned(16))) __bf16 Wtap[32 * KP2];    // per-tap fused weights
    __shared__ float  s_sum[32];
    __shared__ float  s_sq[32];

    const int t   = threadIdx.x;
    const int g   = blockIdx.y;
    const int bb  = blockIdx.z;
    const int ty  = blockIdx.x >> 2;       // 0..31 (row tiles of 2)
    const int tx  = blockIdx.x & 3;        // 0..3  (col tiles of 16)
    const int r0  = ty * 2;
    const int c0  = tx * 16;
    const unsigned wtap_ldsb = (unsigned)(size_t)(&Wtap[0]);

    // stage halo features from bf16 h1 (zero-fill outside image == conv padding)
    for (int idx = t; idx < NHALO * 32; idx += 64) {
        int ch  = idx & 31, pix = idx >> 5;
        int py  = pix / HALO_C, px = pix % HALO_C;
        int y   = r0 - 1 + py, xx = c0 - 1 + px;
        bool inb = (y >= 0) && (y < 64) && (xx >= 0) && (xx < 64);
        float v  = inb ? (float)h1[(bb * COUT + g * 32 + ch) * HW + y * 64 + xx] : 0.0f;
        Flds[pix * KP2 + ch] = (__bf16)(inb ? silu_f(v) : 0.0f);
        float bs[8]; bsplines8(v, bs);
#pragma unroll
        for (int q = 0; q < 8; ++q)
            Flds[pix * KP2 + 32 + ch * 8 + q] = inb ? (__bf16)bs[q] : (__bf16)0.0f;
    }
    for (int idx = t; idx < NHALO * (KP2 - K2); idx += 64)
        Flds[(idx >> 3) * KP2 + K2 + (idx & 7)] = (__bf16)0.0f;
    if (t < 32) { s_sum[t] = 0.0f; s_sq[t] = 0.0f; }

    const int wv_ = t >> 5;   // wave id == local output row (tile is 2x16)
    v8f acc0 = {}; v8f acc1 = {};

    for (int tap = 0; tap < 9; ++tap) {
        __syncthreads();      // previous tap's weights fully consumed; feats staged (tap 0)
        // async DMA this tap's fused weight slice: contiguous 32*KP2 bf16 = 1184 x 16B
        {
            const __bf16* src = Wf2 + (size_t)(g * 9 + tap) * 32 * KP2;
            for (int c = t; c < (32 * KP2 * 2) / 16; c += 64)
                async_copy_b128(wtap_ldsb + c * 16, src + c * 8);
        }
        if (tap < 8) __builtin_prefetch(Wf2 + (size_t)(g * 9 + tap + 1) * 32 * KP2, 0, 1);
        wait_async0();
        __syncthreads();

        const int ky = tap / 3, kx = tap % 3;
        const int rowb = (wv_ + ky) * HALO_C + kx;   // halo pixel base for this wave
#pragma unroll
        for (int kt = 0; kt < 9; ++kt) {
            int kb = kt * 32;
            v16bf bf = load_b_frag(Flds, rowb, kb, KP2);
            v16bf a0 = load_a_frag(Wtap, 0,  kb, KP2);
            v16bf a1 = load_a_frag(Wtap, 16, kb, KP2);
            acc0 = wmma_bf16(a0, bf, acc0);
            acc1 = wmma_bf16(a1, bf, acc1);
        }
    }

    // write h2 + per-channel BN partial sums (lane shuffles -> LDS -> global atomics)
    const int l    = t & 31;
    const int half = l >> 4;
    const int lx   = l & 15;
    const int y    = r0 + wv_;
    const int xx   = c0 + lx;
#pragma unroll
    for (int r = 0; r < 8; ++r) {
        {
            int m  = r + half * 8;
            int ch = g * 32 + m;
            float val = acc0[r] + bb2[ch];
            h2[(bb * COUT + ch) * HW + y * 64 + xx] = val;
            float s = val, q = val * val;
#pragma unroll
            for (int msk = 1; msk < 16; msk <<= 1) {
                s += __shfl_xor(s, msk, 32);
                q += __shfl_xor(q, msk, 32);
            }
            if (lx == 0) { atomicAdd(&s_sum[m], s); atomicAdd(&s_sq[m], q); }
        }
        {
            int m  = r + half * 8 + 16;
            int ch = g * 32 + m;
            float val = acc1[r] + bb2[ch];
            h2[(bb * COUT + ch) * HW + y * 64 + xx] = val;
            float s = val, q = val * val;
#pragma unroll
            for (int msk = 1; msk < 16; msk <<= 1) {
                s += __shfl_xor(s, msk, 32);
                q += __shfl_xor(q, msk, 32);
            }
            if (lx == 0) { atomicAdd(&s_sum[m], s); atomicAdd(&s_sq[m], q); }
        }
    }
    __syncthreads();
    if (t < 32) {
        atomicAdd(&sums[g * 32 + t],  s_sum[t]);
        atomicAdd(&sumsq[g * 32 + t], s_sq[t]);
    }
}

// ---------------- kernel 3: BN finalize (scale/shift per channel) ----------
__global__ void GSC_kan_bnfin(const float* __restrict__ sums, const float* __restrict__ sumsq,
                              const float* __restrict__ gamma, const float* __restrict__ beta,
                              float* __restrict__ scale, float* __restrict__ shift) {
    int ch = blockIdx.x * blockDim.x + threadIdx.x;
    if (ch < COUT) {
        const float inv = 1.0f / 65536.0f;   // N*H*W
        float m  = sums[ch] * inv;
        float v2 = sumsq[ch] * inv - m * m;
        float sc = gamma[ch] * rsqrtf(v2 + 1e-5f);
        scale[ch] = sc;
        shift[ch] = beta[ch] - m * sc;
    }
}

// ---------------- kernel 4: apply BN (vectorized) ----------------
__global__ __launch_bounds__(256) void GSC_kan_bnapply(const float* __restrict__ h2,
                                                       const float* __restrict__ scale,
                                                       const float* __restrict__ shift,
                                                       float* __restrict__ out, int n4) {
    int i = blockIdx.x * 256 + threadIdx.x;
    if (i < n4) {
        int ch = (i >> 10) & 255;            // 1024 float4 per (b,ch) plane
        float4 v = ((const float4*)h2)[i];
        float sc = scale[ch], sh = shift[ch];
        float4 o;
        o.x = v.x * sc + sh; o.y = v.y * sc + sh;
        o.z = v.z * sc + sh; o.w = v.w * sc + sh;
        ((float4*)out)[i] = o;
    }
}

// ---------------- launch ----------------
extern "C" void kernel_launch(void* const* d_in, const int* in_sizes, int n_in,
                              void* d_out, int out_size, void* d_ws, size_t ws_size,
                              hipStream_t stream) {
    (void)in_sizes; (void)n_in; (void)out_size; (void)ws_size;
    const float* x     = (const float*)d_in[0];
    const float* wb1   = (const float*)d_in[1];
    const float* bb1   = (const float*)d_in[2];
    const float* ws1   = (const float*)d_in[3];
    const float* wb2   = (const float*)d_in[4];
    const float* bb2   = (const float*)d_in[5];
    const float* ws2   = (const float*)d_in[6];
    const float* gamma = (const float*)d_in[7];
    const float* beta  = (const float*)d_in[8];

    char* ws = (char*)d_ws;
    __bf16* h1    = (__bf16*)ws;                             // 33,554,432 B
    float*  h2    = (float*)(ws + 33554432);                 // 67,108,864 B
    float*  sums  = (float*)(ws + 100663296);                // 256
    float*  sumsq = sums + 256;                              // 256
    float*  scale = sums + 512;                              // 256
    float*  shift = sums + 768;                              // 256
    __bf16* Wf1   = (__bf16*)(ws + 100667392);               // 86,016 B
    __bf16* Wf2   = (__bf16*)(ws + 100753408);               // 1,363,968 B

    GSC_kan_zero<<<1, 512, 0, stream>>>(sums, 512);
    GSC_kan_prep<<<(NWF2 + 255) / 256, 256, 0, stream>>>(wb1, ws1, wb2, ws2, Wf1, Wf2);

    dim3 g1(NPIX / 64, GRP);
    GSC_kan1<<<g1, 128, 0, stream>>>(x, Wf1, bb1, h1);

    dim3 g2(128, GRP, B_N);
    GSC_kan2<<<g2, 64, 0, stream>>>(h1, Wf2, bb2, h2, sums, sumsq);

    GSC_kan_bnfin<<<1, 256, 0, stream>>>(sums, sumsq, gamma, beta, scale, shift);

    GSC_kan_bnapply<<<16384, 256, 0, stream>>>(h2, scale, shift, (float*)d_out, 4194304);
}
